// SelfAttention_3066606649373
// MI455X (gfx1250) — matrix operand order repaired and checked by reading, not compiled, not run on hardware
//
#include <hip/hip_runtime.h>

typedef __attribute__((ext_vector_type(16))) __bf16 v16bf;
typedef __attribute__((ext_vector_type(8)))  float  v8f;

#define HDIM 768
#define LSEQ 4096
#define NBATCH 2
#define NHEAD 12
#define DHEAD 64
#define LOG2E 1.44269504088896340736f

#if __has_builtin(__builtin_amdgcn_tensor_load_to_lds) && __has_builtin(__builtin_amdgcn_s_wait_tensorcnt)
#define USE_TDM 1
#else
#define USE_TDM 0
#endif

union BF16Frag { v16bf v; uint4 q[2]; unsigned short u[16]; };
union AccFrag  { v8f  v; float f[8]; };

__device__ __forceinline__ unsigned short f2bf(float f) {
    unsigned int u = __float_as_uint(f);
    u += 0x7FFFu + ((u >> 16) & 1u);            // round-to-nearest-even
    return (unsigned short)(u >> 16);
}
__device__ __forceinline__ unsigned int pk2bf(float lo, float hi) {
    return (unsigned int)f2bf(lo) | ((unsigned int)f2bf(hi) << 16);
}
__device__ __forceinline__ float fexp2(float x) {
#if __has_builtin(__builtin_amdgcn_exp2f)
    return __builtin_amdgcn_exp2f(x);           // raw v_exp_f32
#else
    return exp2f(x);
#endif
}

#if USE_TDM
typedef unsigned int v4u  __attribute__((ext_vector_type(4)));
typedef int          v4it __attribute__((ext_vector_type(4)));
typedef int          v8it __attribute__((ext_vector_type(8)));

__device__ __forceinline__ unsigned lds_addr_of(const void* p) {
    return (unsigned)(unsigned long long)p;     // low 32 bits of generic = LDS offset
}
// group1 dword0: workgroup_mask=0 | data_size=2B | pad_enable | pad_interval | pad_amount
#define TDM_D0(padint, padamt) \
    ((1u << 16) | (1u << 20) | ((unsigned)(padint) << 22) | ((unsigned)(padamt) << 25))

// 2D bf16 tile (tile_w elems/row, tile_h rows, global row stride in elems) -> padded LDS.
__device__ __forceinline__ void tdm_load_2d(unsigned lds, const void* gp,
                                            unsigned tile_w, unsigned tile_h,
                                            unsigned row_stride_elts, unsigned d0cfg) {
    const unsigned long long ga = (unsigned long long)gp;
    v4u g0;
    g0[0] = 1u;                                             // count=1, user D#
    g0[1] = lds;                                            // lds_addr
    g0[2] = (unsigned)ga;                                   // global_addr[31:0]
    g0[3] = (unsigned)((ga >> 32) & 0x01FFFFFFu) | 0x80000000u; // addr[56:32] | type=2
    v8it g1;
    g1[0] = (int)d0cfg;
    g1[1] = (int)(tile_w << 16);                            // tensor_dim0 low16 @ [63:48]
    g1[2] = (int)(tile_h << 16);                            // tensor_dim1 low16 @ [95:80]
    g1[3] = (int)(tile_w << 16);                            // tile_dim0 @ [127:112]
    g1[4] = (int)tile_h;                                    // tile_dim1 @ [143:128]
    g1[5] = (int)row_stride_elts;                           // tensor_dim0_stride low32
    g1[6] = 0;
    g1[7] = 0;
    v4it z4 = {0, 0, 0, 0};
#if __clang_major__ >= 23
    v8it z8 = {0, 0, 0, 0, 0, 0, 0, 0};
    __builtin_amdgcn_tensor_load_to_lds(g0, g1, z4, z4, z8, 0);
#else
    __builtin_amdgcn_tensor_load_to_lds(g0, g1, z4, z4, 0);
#endif
}
#endif  // USE_TDM

// ---- 16-lane butterfly reductions (C/D rows live in 16-lane halves)
#if __has_builtin(__builtin_amdgcn_update_dpp)
#define DPP_XF(x, ctrl) \
    __int_as_float(__builtin_amdgcn_update_dpp(0, __float_as_int(x), (ctrl), 0xF, 0xF, true))
__device__ __forceinline__ float rmax16(float x) {
    x = fmaxf(x, DPP_XF(x, 0xB1));
    x = fmaxf(x, DPP_XF(x, 0x4E));
    x = fmaxf(x, DPP_XF(x, 0x141));
    x = fmaxf(x, DPP_XF(x, 0x140));
    return x;
}
__device__ __forceinline__ float rsum16(float x) {
    x += DPP_XF(x, 0xB1);
    x += DPP_XF(x, 0x4E);
    x += DPP_XF(x, 0x141);
    x += DPP_XF(x, 0x140);
    return x;
}
#else
__device__ __forceinline__ float rmax16(float x) {
    x = fmaxf(x, __shfl_xor(x, 1, 32));
    x = fmaxf(x, __shfl_xor(x, 2, 32));
    x = fmaxf(x, __shfl_xor(x, 4, 32));
    x = fmaxf(x, __shfl_xor(x, 8, 32));
    return x;
}
__device__ __forceinline__ float rsum16(float x) {
    x += __shfl_xor(x, 1, 32);
    x += __shfl_xor(x, 2, 32);
    x += __shfl_xor(x, 4, 32);
    x += __shfl_xor(x, 8, 32);
    return x;
}
#endif

// A fragment (16x32 bf16) from row-major LDS tile: two contiguous b128 reads.
__device__ __forceinline__ v16bf load_a_frag(const unsigned short* t, int stride, int lane) {
    const int m  = lane & 15;
    const int kb = (lane >> 4) << 3;
    const unsigned short* r = t + m * stride + kb;
    BF16Frag f;
    f.q[0] = *(const uint4*)(r);
    f.q[1] = *(const uint4*)(r + 16);
    return f.v;
}

// B fragment (32x16 bf16) from N-major tile t[n][k]: two contiguous b128 reads.
__device__ __forceinline__ v16bf load_b_frag(const unsigned short* t, int stride,
                                             int rowbase, int colbase, int lane) {
    const int n  = lane & 15;
    const int kb = (lane >> 4) << 3;
    const unsigned short* r = t + (rowbase + n) * stride + colbase + kb;
    BF16Frag f;
    f.q[0] = *(const uint4*)(r);
    f.q[1] = *(const uint4*)(r + 16);
    return f.v;
}

// ---------------------------------------------------------------------------
// One-shot converters (all fp32->bf16 VALU kept out of the hot GEMM loops).
// ---------------------------------------------------------------------------
__global__ __launch_bounds__(256)
void f32_to_bf16(const float* __restrict__ in, unsigned short* __restrict__ out, int n8) {
    int i = blockIdx.x * 256 + threadIdx.x;
    if (i < n8) {
        const float4 a = ((const float4*)in)[2 * i];
        const float4 b = ((const float4*)in)[2 * i + 1];
        uint4 o;
        o.x = pk2bf(a.x, a.y); o.y = pk2bf(a.z, a.w);
        o.z = pk2bf(b.x, b.y); o.w = pk2bf(b.z, b.w);
        ((uint4*)out)[i] = o;
    }
}

__global__ __launch_bounds__(256)
void w_to_bf16t(const float* __restrict__ W, unsigned short* __restrict__ Wt) {
    __shared__ __align__(16) unsigned short ls[64 * 72];
    const int k0 = blockIdx.x * 64, n0 = blockIdx.y * 64;
    const int tid = threadIdx.x;
#pragma unroll
    for (int i = 0; i < 4; ++i) {
        int s = tid + i * 256;
        int r = s >> 4, c4 = (s & 15) << 2;
        const float4 d = *(const float4*)(W + (size_t)(k0 + r) * HDIM + n0 + c4);
        ls[(c4 + 0) * 72 + r] = f2bf(d.x);
        ls[(c4 + 1) * 72 + r] = f2bf(d.y);
        ls[(c4 + 2) * 72 + r] = f2bf(d.z);
        ls[(c4 + 3) * 72 + r] = f2bf(d.w);
    }
    __syncthreads();
#pragma unroll
    for (int i = 0; i < 2; ++i) {
        int s = tid + i * 256;
        int r = s >> 3, c8 = (s & 7) << 3;
        *(uint4*)(Wt + (size_t)(n0 + r) * HDIM + k0 + c8) = *(const uint4*)(ls + r * 72 + c8);
    }
}

// ---------------------------------------------------------------------------
// GEMM: Y = (X(8192x768,bf16) @ Wt^T + b) * oscale; Wt bf16 [n][k].
// 256 thr = 8 waves; tile 128x64; wave 32x32; double-buffered; TDM staging.
// ---------------------------------------------------------------------------
template <bool OUT_ATTN>
__global__ __launch_bounds__(256)
void gemm768(const unsigned short* __restrict__ X, const unsigned short* __restrict__ Wt,
             const float* __restrict__ bias, void* __restrict__ Yv, float oscale) {
    __shared__ __align__(16) unsigned short lsA[2][128 * 48];   // 128 rows x 32 K (+pad)
    __shared__ __align__(16) unsigned short lsBt[2][64 * 40];   // 64 cols x 32 K (+pad)

    const int tid  = threadIdx.x;
    const int lane = tid & 31;
    const int wv   = tid >> 5;
    const int wm   = wv >> 1;
    const int wn   = wv & 1;
    const int gm0  = blockIdx.x * 128;
    const int gn0  = blockIdx.y * 64;
    const int NK   = HDIM / 32;                 // 24

    auto stage = [&](int kt) {
        const int k0 = kt * 32;
#if USE_TDM
        if (wv == 0) {
            // A: rows of 64B + 32B pad (interval code 3, amount code 7) -> stride 48 shorts
            tdm_load_2d(lds_addr_of(lsA[kt & 1]),
                        X + (size_t)gm0 * HDIM + k0, 32, 128, HDIM, TDM_D0(3, 7));
            // B: rows of 64B + 16B pad (interval code 3, amount code 3) -> stride 40 shorts
            tdm_load_2d(lds_addr_of(lsBt[kt & 1]),
                        Wt + (size_t)gn0 * HDIM + k0, 32, 64, HDIM, TDM_D0(3, 3));
        }
#else
        unsigned short* A = lsA[kt & 1];
        unsigned short* B = lsBt[kt & 1];
#pragma unroll
        for (int i = 0; i < 2; ++i) {
            int s = tid + i * 256;
            int row = s >> 2, c8 = (s & 3) << 3;
            *(uint4*)(A + row * 48 + c8) =
                *(const uint4*)(X + (size_t)(gm0 + row) * HDIM + k0 + c8);
        }
        {
            int r = tid >> 2, c8 = (tid & 3) << 3;
            *(uint4*)(B + r * 40 + c8) =
                *(const uint4*)(Wt + (size_t)(gn0 + r) * HDIM + k0 + c8);
        }
#endif
        if (kt + 2 < NK) {                      // prefetch 2 tiles ahead into GL2
            const int kp0 = (kt + 2) * 32;
            if (tid < 128)
                __builtin_prefetch(X + (size_t)(gm0 + tid) * HDIM + kp0, 0, 0);
            else if (tid - 128 < 64)
                __builtin_prefetch(Wt + (size_t)(gn0 + tid - 128) * HDIM + kp0, 0, 0);
        }
    };

    AccFrag acc[2][2];
#pragma unroll
    for (int i = 0; i < 2; ++i)
#pragma unroll
        for (int j = 0; j < 2; ++j)
#pragma unroll
            for (int e = 0; e < 8; ++e) acc[i][j].f[e] = 0.0f;

    stage(0);
    for (int kt = 0; kt < NK; ++kt) {
        __syncthreads();
        if (kt + 1 < NK) stage(kt + 1);
#if USE_TDM
        if (kt + 1 < NK) __builtin_amdgcn_s_wait_tensorcnt(2);  // tile kt done, kt+1 in flight
        else             __builtin_amdgcn_s_wait_tensorcnt(0);
        __syncthreads();                        // publish TDM-written LDS to all waves
#endif
        const unsigned short* A = lsA[kt & 1];
        const unsigned short* B = lsBt[kt & 1];
        v16bf a0 = load_a_frag(A + (wm * 32 + 0)  * 48, 48, lane);
        v16bf a1 = load_a_frag(A + (wm * 32 + 16) * 48, 48, lane);
        v16bf b0 = load_b_frag(B, 40, wn * 32 + 0,  0, lane);
        v16bf b1 = load_b_frag(B, 40, wn * 32 + 16, 0, lane);

        acc[0][0].v = __builtin_amdgcn_wmma_f32_16x16x32_bf16(false, a0, false, b0, (short)0, acc[0][0].v, false, false);
        acc[0][1].v = __builtin_amdgcn_wmma_f32_16x16x32_bf16(false, a0, false, b1, (short)0, acc[0][1].v, false, false);
        acc[1][0].v = __builtin_amdgcn_wmma_f32_16x16x32_bf16(false, a1, false, b0, (short)0, acc[1][0].v, false, false);
        acc[1][1].v = __builtin_amdgcn_wmma_f32_16x16x32_bf16(false, a1, false, b1, (short)0, acc[1][1].v, false, false);
    }

    const int cn = lane & 15;
    const int rb = (lane >> 4) << 3;
#pragma unroll
    for (int mi = 0; mi < 2; ++mi)
#pragma unroll
        for (int ni = 0; ni < 2; ++ni) {
            const int col  = gn0 + wn * 32 + ni * 16 + cn;
            const float bc = bias[col];
#pragma unroll
            for (int j = 0; j < 8; ++j) {
                const int row = gm0 + wm * 32 + mi * 16 + rb + j;
                const float val = (acc[mi][ni].f[j] + bc) * oscale;
                if (OUT_ATTN) {
                    const int bn = row >> 12, l = row & (LSEQ - 1);
                    const int h  = col >> 6,  d = col & (DHEAD - 1);
                    ((unsigned short*)Yv)[(((size_t)(bn * NHEAD + h) * LSEQ) + l) * DHEAD + d] = f2bf(val);
                } else {
                    ((float*)Yv)[(size_t)row * HDIM + col] = val;
                }
            }
        }
}

// ---------------------------------------------------------------------------
// Flash attention: block = 128-row Q tile of one (batch, head); 256 thr = 8
// waves, 16 query rows each; 64-key double-buffered K/V blocks; K via TDM.
// Q pre-scaled by log2(e)/sqrt(D); base-2 softmax (raw v_exp_f32).
// ---------------------------------------------------------------------------
__global__ __launch_bounds__(256)
void flash_attn(const unsigned short* __restrict__ Q, const unsigned short* __restrict__ K,
                const unsigned short* __restrict__ V, unsigned short* __restrict__ O) {
    __shared__ __align__(16) unsigned short lsK[2][64 * 72];    // [key][d] (+pad)
    __shared__ __align__(16) unsigned short lsVt[2][64 * 72];   // [d][key] (+pad)
    __shared__ __align__(16) unsigned short lsP[8][16 * 72];    // per-wave P tile 16x64

    const int tid  = threadIdx.x;
    const int lane = tid & 31;
    const int wv   = tid >> 5;
    const int q0   = blockIdx.x * 128;
    const int h    = blockIdx.y;
    const int bn   = blockIdx.z;
    const size_t headbase = (size_t)(bn * NHEAD + h) * LSEQ * DHEAD;
    const int qrow = q0 + wv * 16;
    const int NKT  = LSEQ / 64;                 // 64

    auto stage = [&](int kt) {
        const int kb0 = kt * 64;
#if USE_TDM
        if (wv == 0) {
            // K: rows of 128B + 16B pad (interval code 4, amount code 3) -> stride 72 shorts
            tdm_load_2d(lds_addr_of(lsK[kt & 1]),
                        K + headbase + (size_t)kb0 * DHEAD, 64, 64, DHEAD, TDM_D0(4, 3));
        }
#else
        unsigned short* Kb = lsK[kt & 1];
#pragma unroll
        for (int i = 0; i < 2; ++i) {
            int s = tid + i * 256;
            int row = s >> 3, c8 = (s & 7) << 3;
            *(uint4*)(Kb + row * 72 + c8) =
                *(const uint4*)(K + headbase + (size_t)(kb0 + row) * DHEAD + c8);
        }
#endif
        {   // V transposed, pair-packed: 256 slots (TDM cannot transpose).
            unsigned short* Vt = lsVt[kt & 1];
            int kp = tid >> 3, dq = tid & 7;
            int key = kp * 2, db = dq * 8;
            uint4 va = *(const uint4*)(V + headbase + (size_t)(kb0 + key)     * DHEAD + db);
            uint4 vb = *(const uint4*)(V + headbase + (size_t)(kb0 + key + 1) * DHEAD + db);
            const unsigned short* pa = (const unsigned short*)&va;
            const unsigned short* pb = (const unsigned short*)&vb;
#pragma unroll
            for (int i = 0; i < 8; ++i)
                *(unsigned int*)(Vt + (db + i) * 72 + key) =
                    (unsigned int)pa[i] | ((unsigned int)pb[i] << 16);
        }
        if (kt + 2 < NKT) {
            const size_t nb = headbase + (size_t)(kt + 2) * 64 * DHEAD;
            __builtin_prefetch(K + nb + (size_t)tid * 16, 0, 0);
            __builtin_prefetch(V + nb + (size_t)tid * 16, 0, 0);
        }
    };

    // Q A-fragments (16 rows x 64 d) in registers for the whole sweep.
    BF16Frag aq[2];
    {
        const int m = lane & 15, kb = (lane >> 4) << 3;
        const unsigned short* qp = Q + headbase + (size_t)(qrow + m) * DHEAD;
#pragma unroll
        for (int kc = 0; kc < 2; ++kc) {
            aq[kc].q[0] = *(const uint4*)(qp + kc * 32 + kb);
            aq[kc].q[1] = *(const uint4*)(qp + kc * 32 + kb + 16);
        }
    }

    float mrow[8], lrow[8];
    AccFrag o[4];
#pragma unroll
    for (int j = 0; j < 8; ++j) { mrow[j] = -1.0e30f; lrow[j] = 0.0f; }
#pragma unroll
    for (int t = 0; t < 4; ++t)
#pragma unroll
        for (int j = 0; j < 8; ++j) o[t].f[j] = 0.0f;

    stage(0);
    for (int kt = 0; kt < NKT; ++kt) {
        __syncthreads();
        if (kt + 1 < NKT) stage(kt + 1);
#if USE_TDM
        if (kt + 1 < NKT) __builtin_amdgcn_s_wait_tensorcnt(1);  // K(kt) done
        else             __builtin_amdgcn_s_wait_tensorcnt(0);
        __syncthreads();
#endif
        const unsigned short* Kb = lsK[kt & 1];
        const unsigned short* Vt = lsVt[kt & 1];

        // S = Q * K^T over four 16-key groups (fp32 accumulate, base-2 logits).
        float sv[4][8];
#pragma unroll
        for (int g = 0; g < 4; ++g) {
            AccFrag s;
#pragma unroll
            for (int j = 0; j < 8; ++j) s.f[j] = 0.0f;
#pragma unroll
            for (int kc = 0; kc < 2; ++kc) {
                v16bf b = load_b_frag(Kb, 72, g * 16, kc * 32, lane);
                s.v = __builtin_amdgcn_wmma_f32_16x16x32_bf16(false, aq[kc].v, false, b, (short)0, s.v, false, false);
            }
#pragma unroll
            for (int j = 0; j < 8; ++j) sv[g][j] = s.f[j];
        }

        // Online softmax in base 2.
        float alpha[8];
#pragma unroll
        for (int j = 0; j < 8; ++j) {
            float t = fmaxf(fmaxf(sv[0][j], sv[1][j]), fmaxf(sv[2][j], sv[3][j]));
            t = rmax16(t);
            const float mn = fmaxf(mrow[j], t);
            alpha[j] = fexp2(mrow[j] - mn);
            mrow[j]  = mn;
            float rs = 0.0f;
#pragma unroll
            for (int g = 0; g < 4; ++g) {
                const float p = fexp2(sv[g][j] - mn);
                sv[g][j] = p;
                rs += p;
            }
            lrow[j] = lrow[j] * alpha[j] + rsum16(rs);
        }
#pragma unroll
        for (int t = 0; t < 4; ++t)
#pragma unroll
            for (int j = 0; j < 8; ++j) o[t].f[j] *= alpha[j];

        // P: C-layout -> A-layout via per-wave LDS round-trip (16x64 bf16).
        {
            const int cn = lane & 15, rb = (lane >> 4) << 3;
            unsigned short* pp = lsP[wv];
#pragma unroll
            for (int g = 0; g < 4; ++g)
#pragma unroll
                for (int j = 0; j < 8; ++j)
                    pp[(rb + j) * 72 + g * 16 + cn] = f2bf(sv[g][j]);
        }
        v16bf ap0 = load_a_frag(lsP[wv] + 0,  72, lane);
        v16bf ap1 = load_a_frag(lsP[wv] + 32, 72, lane);

        // O += P * V (4 d-tiles x 2 K-chunks) from transposed V tile.
#pragma unroll
        for (int t = 0; t < 4; ++t) {
            v16bf bv0 = load_b_frag(Vt, 72, t * 16, 0,  lane);
            v16bf bv1 = load_b_frag(Vt, 72, t * 16, 32, lane);
            o[t].v = __builtin_amdgcn_wmma_f32_16x16x32_bf16(false, ap0, false, bv0, (short)0, o[t].v, false, false);
            o[t].v = __builtin_amdgcn_wmma_f32_16x16x32_bf16(false, ap1, false, bv1, (short)0, o[t].v, false, false);
        }
    }

    // Normalize and write attention output, (n, l, hd) row-major bf16.
    const int cn = lane & 15, rb = (lane >> 4) << 3;
    float inv[8];
#pragma unroll
    for (int j = 0; j < 8; ++j) inv[j] = 1.0f / lrow[j];
#pragma unroll
    for (int t = 0; t < 4; ++t)
#pragma unroll
        for (int j = 0; j < 8; ++j) {
            const int row = qrow + rb + j;
            const int col = h * DHEAD + t * 16 + cn;
            O[(size_t)(bn * LSEQ + row) * HDIM + col] = f2bf(o[t].f[j] * inv[j]);
        }
}

extern "C" void kernel_launch(void* const* d_in, const int* in_sizes, int n_in,
                              void* d_out, int out_size, void* d_ws, size_t ws_size,
                              hipStream_t stream) {
    (void)in_sizes; (void)n_in; (void)out_size; (void)ws_size;
    const float* x  = (const float*)d_in[0];
    const float* Wq = (const float*)d_in[1];
    const float* bq = (const float*)d_in[2];
    const float* Wk = (const float*)d_in[3];
    const float* bk = (const float*)d_in[4];
    const float* Wv = (const float*)d_in[5];
    const float* bv = (const float*)d_in[6];
    const float* Wo = (const float*)d_in[7];
    const float* bo = (const float*)d_in[8];
    float* out = (float*)d_out;

    const size_t per = (size_t)NBATCH * NHEAD * LSEQ * DHEAD;   // 6,291,456
    const size_t wsz = (size_t)HDIM * HDIM;                     //   589,824
    unsigned short* xb  = (unsigned short*)d_ws;
    unsigned short* Wtq = xb  + per;
    unsigned short* Wtk = Wtq + wsz;
    unsigned short* Wtv = Wtk + wsz;
    unsigned short* Wto = Wtv + wsz;
    unsigned short* Qb  = Wto + wsz;
    unsigned short* Kb  = Qb + per;
    unsigned short* Vb  = Kb + per;
    unsigned short* Ab  = Vb + per;

    f32_to_bf16<<<(int)(per / (256 * 8)), 256, 0, stream>>>(x, xb, (int)(per / 8));
    dim3 gt(HDIM / 64, HDIM / 64, 1);
    w_to_bf16t<<<gt, 256, 0, stream>>>(Wq, Wtq);
    w_to_bf16t<<<gt, 256, 0, stream>>>(Wk, Wtk);
    w_to_bf16t<<<gt, 256, 0, stream>>>(Wv, Wtv);
    w_to_bf16t<<<gt, 256, 0, stream>>>(Wo, Wto);

    dim3 gg(64, 12, 1);
    const float qscale = 0.125f * LOG2E;   // 1/sqrt(D) * log2(e): base-2 softmax
    gemm768<true><<<gg, 256, 0, stream>>>(xb, Wtq, bq, (void*)Qb, qscale);
    gemm768<true><<<gg, 256, 0, stream>>>(xb, Wtk, bk, (void*)Kb, 1.0f);
    gemm768<true><<<gg, 256, 0, stream>>>(xb, Wtv, bv, (void*)Vb, 1.0f);

    dim3 ga(LSEQ / 128, NHEAD, NBATCH);
    flash_attn<<<ga, 256, 0, stream>>>(Qb, Kb, Vb, Ab);

    gemm768<false><<<gg, 256, 0, stream>>>(Ab, Wto, bo, (void*)out, 1.0f);
}
